// SpaTB_30571577213036
// MI455X (gfx1250) — matrix operand order repaired and checked by reading, not compiled, not run on hardware
//
#include <hip/hip_runtime.h>
#include <hip/hip_bf16.h>

typedef __attribute__((ext_vector_type(16))) __bf16 v16bf;
typedef __attribute__((ext_vector_type(8)))  float  v8f;
typedef __attribute__((vector_size(16))) int v4i;
typedef __attribute__((vector_size(8)))  int v2i;

#define TOK   32768
#define DIM   192
#define HEADS 6
#define HD    32
#define MLP_H 768

#define WMMA_BF16(a,b,c) __builtin_amdgcn_wmma_f32_16x16x32_bf16(false,(a),false,(b),(short)0,(c),false,false)

// ---- async global->LDS path (CDNA5), guarded with fallback ----------------
#if defined(__has_builtin)
# if __has_builtin(__builtin_amdgcn_global_load_async_to_lds_b128) && \
     __has_builtin(__builtin_amdgcn_global_load_async_to_lds_b64)
#  define ASYNC_LDS 1
# endif
#endif
#ifndef ASYNC_LDS
# define ASYNC_LDS 0
#endif

static __device__ __forceinline__ void cp_lds16(void* lds, const void* g) {
#if ASYNC_LDS
  __builtin_amdgcn_global_load_async_to_lds_b128(
      (__attribute__((address_space(1))) v4i*)g,
      (__attribute__((address_space(3))) v4i*)lds, 0, 0);
#else
  *reinterpret_cast<uint4*>(lds) = *reinterpret_cast<const uint4*>(g);
#endif
}
static __device__ __forceinline__ void cp_lds8(void* lds, const void* g) {
#if ASYNC_LDS
  __builtin_amdgcn_global_load_async_to_lds_b64(
      (__attribute__((address_space(1))) v2i*)g,
      (__attribute__((address_space(3))) v2i*)lds, 0, 0);
#else
  *reinterpret_cast<uint2*>(lds) = *reinterpret_cast<const uint2*>(g);
#endif
}
static __device__ __forceinline__ void async_wait() {
#if ASYNC_LDS
  asm volatile("s_wait_asynccnt 0x0" ::: "memory");
#endif
}

// ---- WMMA fragment loaders (contiguous k order, k = 16*half + j, wave32) -------------------
// A (16x32 bf16): lanes 0-15 row=r K={0..15}; lanes 16-31 K={16..31}
static __device__ __forceinline__ v16bf load_a_frag(const __bf16* A, int stride,
                                                    int row0, int r, int half) {
  const __bf16* p = A + (size_t)(row0 + r) * stride + half * 16;
  v16bf a;
#pragma unroll
  for (int j = 0; j < 8; ++j) a[j] = p[j];
#pragma unroll
  for (int j = 0; j < 8; ++j) a[8 + j] = p[8 + j];
  return a;
}
// B (32x16 bf16), B stored [k][n]: lane col = r, k = half*16 + j
static __device__ __forceinline__ v16bf load_b_frag(const __bf16* B, int stride,
                                                    int col, int r, int half) {
  v16bf b;
#pragma unroll
  for (int j = 0; j < 16; ++j) b[j] = B[(size_t)(half * 16 + j) * stride + col + r];
  return b;
}
// B for Q·K^T: K stored [key][d]; B[k=d][n=key] => contiguous row read
static __device__ __forceinline__ v16bf load_bt_frag(const __bf16* Kb, int stride,
                                                     int col, int r, int half) {
  const __bf16* p = Kb + (size_t)(col + r) * stride + half * 16;
  v16bf b;
#pragma unroll
  for (int j = 0; j < 16; ++j) b[j] = p[j];
  return b;
}

// ---- one-time converters ---------------------------------------------------
__global__ void __launch_bounds__(256) f32_to_bf16_kernel(const float* __restrict__ in,
                                                          __bf16* __restrict__ out, int n) {
  int i = blockIdx.x * 256 + threadIdx.x;
  if (i < n) out[i] = (__bf16)in[i];
}

// bias_t[((head*576)+key)*256 + qr] = rpb[rpi[qr*576+key]*HEADS + head]
__global__ void __launch_bounds__(256) bias_pre_kernel(const int* __restrict__ rpi,
                                                       const float* __restrict__ rpb,
                                                       float* __restrict__ bias_t) {
  int idx = blockIdx.x * 256 + threadIdx.x;   // 6*576*256 total
  int qr = idx & 255;
  int rest = idx >> 8;
  int key = rest % 576;
  int head = rest / 576;
  bias_t[idx] = rpb[rpi[qr * 576 + key] * HEADS + head];
}

// ---- LayerNorm: one wave per 192-elem row, output bf16 --------------------
__global__ void __launch_bounds__(256) ln_bf16_kernel(const float* __restrict__ x,
                                                      const float* __restrict__ g,
                                                      const float* __restrict__ bta,
                                                      __bf16* __restrict__ out) {
  const int wave = threadIdx.x >> 5, lane = threadIdx.x & 31;
  const int row = blockIdx.x * 8 + wave;
  const float* xr = x + (size_t)row * DIM;
  float v[6], s = 0.f, sq = 0.f;
#pragma unroll
  for (int k = 0; k < 6; ++k) { v[k] = xr[lane + 32 * k]; s += v[k]; sq += v[k] * v[k]; }
#pragma unroll
  for (int m = 1; m < 32; m <<= 1) { s += __shfl_xor(s, m, 32); sq += __shfl_xor(sq, m, 32); }
  const float mean = s * (1.f / DIM);
  const float var  = sq * (1.f / DIM) - mean * mean;
  const float rs   = rsqrtf(var + 1e-5f);
  __bf16* orow = out + (size_t)row * DIM;
#pragma unroll
  for (int k = 0; k < 6; ++k) {
    int c = lane + 32 * k;
    orow[c] = (__bf16)((v[k] - mean) * rs * g[c] + bta[c]);
  }
}

// ---- Generic bf16 WMMA GEMM: 64x64 tile / block, 8 waves, double-buffered -
// EPI 0: out=bf16 acc+bias ; 1: out=bf16 gelu(acc+bias) ; 2: out=f32 acc+bias+resid
template <int EPI>
__global__ void __launch_bounds__(256)
gemm_bf16_kernel(const __bf16* __restrict__ A, int lda,
                 const __bf16* __restrict__ Wb, int ldw,
                 const float* __restrict__ bias,
                 void* __restrict__ OutP, int ldo,
                 const float* __restrict__ resid, int K) {
  __shared__ __bf16 As[2][64][48];
  __shared__ __bf16 Bs[2][32][80];
  const int tid = threadIdx.x, wave = tid >> 5, lane = tid & 31;
  const int half = lane >> 4, r = lane & 15;
  const int m0 = blockIdx.x * 64, n0 = blockIdx.y * 64;
  const int mt = wave >> 1, ntb = (wave & 1) * 2;
  v8f acc[2];
#pragma unroll
  for (int t = 0; t < 2; ++t)
#pragma unroll
    for (int i = 0; i < 8; ++i) acc[t][i] = 0.f;

  const int arow = tid >> 2, acb = (tid & 3) * 8;
  const int brow = tid >> 3, bcb = (tid & 7) * 8;
  const __bf16* Ag = A + (size_t)(m0 + arow) * lda + acb;
  const __bf16* Bg = Wb + (size_t)brow * ldw + n0 + bcb;

  cp_lds16(&As[0][arow][acb], Ag);
  cp_lds16(&Bs[0][brow][bcb], Bg);
  async_wait();
  __syncthreads();

  const int nk = K >> 5;
  for (int k = 0; k < nk; ++k) {
    if (k + 1 < nk) { // stage next tile into the other buffer while computing
      cp_lds16(&As[(k + 1) & 1][arow][acb], Ag + (size_t)(k + 1) * 32);
      cp_lds16(&Bs[(k + 1) & 1][brow][bcb], Bg + (size_t)(k + 1) * 32 * ldw);
    }
    const __bf16* as = &As[k & 1][0][0];
    const __bf16* bs = &Bs[k & 1][0][0];
    v16bf a  = load_a_frag(as, 48, mt * 16, r, half);
    v16bf b0 = load_b_frag(bs, 80, ntb * 16, r, half);
    v16bf b1 = load_b_frag(bs, 80, (ntb + 1) * 16, r, half);
    acc[0] = WMMA_BF16(a, b0, acc[0]);
    acc[1] = WMMA_BF16(a, b1, acc[1]);
    if (k + 1 < nk) { async_wait(); __syncthreads(); }
  }
#pragma unroll
  for (int t = 0; t < 2; ++t)
#pragma unroll
    for (int i = 0; i < 8; ++i) {
      const int row = m0 + mt * 16 + i + 8 * half;
      const int col = n0 + (ntb + t) * 16 + r;
      float val = acc[t][i] + bias[col];
      if (EPI == 0) {
        ((__bf16*)OutP)[(size_t)row * ldo + col] = (__bf16)val;
      } else if (EPI == 1) {
        val = 0.5f * val * (1.f + erff(val * 0.70710678118654752f));
        ((__bf16*)OutP)[(size_t)row * ldo + col] = (__bf16)val;
      } else {
        val += resid[(size_t)row * ldo + col];
        ((float*)OutP)[(size_t)row * ldo + col] = val;
      }
    }
}

// ---- Attention: block = (window, head), 512 thr / 16 waves, flash over 9x64
__global__ void __launch_bounds__(512)
attn_kernel(const __bf16* __restrict__ qbf, const __bf16* __restrict__ kvbf,
            const float* __restrict__ bias_t, __bf16* __restrict__ obf) {
  // smemA: q staging (256 x stride40 = 10240 elems), then per-wave P scratch.
  __shared__ __bf16 smemA[16 * 16 * 72];
  __shared__ __bf16 ks[2][64][40];
  __shared__ __bf16 vs[2][64][40];
  const int win = blockIdx.x, head = blockIdx.y;
  const int bb = win >> 6, wr = (win >> 3) & 7, wc = win & 7;
  const int tid = threadIdx.x, wave = tid >> 5, lane = tid & 31;
  const int half = lane >> 4, r = lane & 15;
  const int kvrow = tid >> 3, kvcb = (tid & 7) * 4;

  { // q window (256 x 32) -> LDS, half row per thread
    const int qi = tid >> 1, cb = (tid & 1) * 16;
    const int i = qi >> 4, j = qi & 15;
    const int tok = bb * 16384 + (wr * 16 + i) * 128 + (wc * 16 + j);
    const __bf16* src = qbf + (size_t)tok * DIM + head * HD + cb;
    cp_lds16(&smemA[qi * 40 + cb], src);
    cp_lds16(&smemA[qi * 40 + cb + 8], src + 8);
  }
  // stage k/v chunk 0; zero-fill the padded (out-of-image) rows
  {
    const int kidx = kvrow;           // ch = 0
    const int i2 = kidx / 24, j2 = kidx % 24;
    const int gi = wr * 16 - 4 + i2, gj = wc * 16 - 4 + j2;
    if (gi >= 0 && gi < 128 && gj >= 0 && gj < 128) {
      const int tok = bb * 16384 + gi * 128 + gj;
      const __bf16* kp = kvbf + (size_t)tok * (2 * DIM) + head * HD + kvcb;
      cp_lds8(&ks[0][kvrow][kvcb], kp);
      cp_lds8(&vs[0][kvrow][kvcb], kp + DIM);
    } else {
      uint2 z = {0u, 0u};
      *reinterpret_cast<uint2*>(&ks[0][kvrow][kvcb]) = z;
      *reinterpret_cast<uint2*>(&vs[0][kvrow][kvcb]) = z;
    }
  }
  async_wait();
  __syncthreads();
  const v16bf aq = load_a_frag(smemA, 40, wave * 16, r, half);
  __syncthreads();   // all q-frag reads done before smemA becomes P scratch
  __bf16* pb_ = smemA + wave * (16 * 72);

  float mrow[8], lrow[8];
  v8f oacc[2];
#pragma unroll
  for (int i = 0; i < 8; ++i) { mrow[i] = -1e30f; lrow[i] = 0.f; }
#pragma unroll
  for (int dt = 0; dt < 2; ++dt)
#pragma unroll
    for (int i = 0; i < 8; ++i) oacc[dt][i] = 0.f;

  const float scale = 0.17677669529663689f; // 1/sqrt(32)
  const float* bias_base = bias_t + (size_t)head * 576 * 256 + wave * 16 + 8 * half;

  for (int ch = 0; ch < 9; ++ch) {
    const int buf = ch & 1;
    if (ch + 1 < 9) { // stage next k/v chunk while computing this one
      const int kidx = (ch + 1) * 64 + kvrow;
      const int i2 = kidx / 24, j2 = kidx % 24;
      const int gi = wr * 16 - 4 + i2, gj = wc * 16 - 4 + j2;
      if (gi >= 0 && gi < 128 && gj >= 0 && gj < 128) {
        const int tok = bb * 16384 + gi * 128 + gj;
        const __bf16* kp = kvbf + (size_t)tok * (2 * DIM) + head * HD + kvcb;
        cp_lds8(&ks[buf ^ 1][kvrow][kvcb], kp);
        cp_lds8(&vs[buf ^ 1][kvrow][kvcb], kp + DIM);
      } else {
        uint2 z = {0u, 0u};
        *reinterpret_cast<uint2*>(&ks[buf ^ 1][kvrow][kvcb]) = z;
        *reinterpret_cast<uint2*>(&vs[buf ^ 1][kvrow][kvcb]) = z;
      }
    }
    const __bf16* ksb = &ks[buf][0][0];
    const __bf16* vsb = &vs[buf][0][0];

    // scores: 4 x (16q x 16k) WMMA
    v8f sc[4];
#pragma unroll
    for (int nt = 0; nt < 4; ++nt) {
      v16bf bk = load_bt_frag(ksb, 40, nt * 16, r, half);
      v8f z;
#pragma unroll
      for (int i = 0; i < 8; ++i) z[i] = 0.f;
      sc[nt] = WMMA_BF16(aq, bk, z);
    }
    // scale + precomputed bias (vector loads), chunk max
    float cmax[8];
#pragma unroll
    for (int i = 0; i < 8; ++i) cmax[i] = -1e30f;
#pragma unroll
    for (int nt = 0; nt < 4; ++nt) {
      const int key = ch * 64 + nt * 16 + r;
      const float* bp = bias_base + (size_t)key * 256;
      float4 b0 = *reinterpret_cast<const float4*>(bp);
      float4 b1 = *reinterpret_cast<const float4*>(bp + 4);
      float bv[8] = {b0.x, b0.y, b0.z, b0.w, b1.x, b1.y, b1.z, b1.w};
#pragma unroll
      for (int i = 0; i < 8; ++i) {
        float sv = sc[nt][i] * scale + bv[i];
        sc[nt][i] = sv;
        cmax[i] = fmaxf(cmax[i], sv);
      }
    }
#pragma unroll
    for (int m = 1; m < 16; m <<= 1)
#pragma unroll
      for (int i = 0; i < 8; ++i) cmax[i] = fmaxf(cmax[i], __shfl_xor(cmax[i], m, 32));
    float alpha[8], psum[8];
#pragma unroll
    for (int i = 0; i < 8; ++i) {
      const float mn = fmaxf(mrow[i], cmax[i]);
      alpha[i] = __expf(mrow[i] - mn);
      mrow[i] = mn;
      psum[i] = 0.f;
    }
#pragma unroll
    for (int nt = 0; nt < 4; ++nt)
#pragma unroll
      for (int i = 0; i < 8; ++i) {
        const float p = __expf(sc[nt][i] - mrow[i]);
        sc[nt][i] = p;
        psum[i] += p;
      }
#pragma unroll
    for (int m = 1; m < 16; m <<= 1)
#pragma unroll
      for (int i = 0; i < 8; ++i) psum[i] += __shfl_xor(psum[i], m, 32);
#pragma unroll
    for (int i = 0; i < 8; ++i) lrow[i] = lrow[i] * alpha[i] + psum[i];
#pragma unroll
    for (int dt = 0; dt < 2; ++dt)
#pragma unroll
      for (int i = 0; i < 8; ++i) oacc[dt][i] *= alpha[i];
    // C-layout -> A-layout via per-wave LDS scratch
#pragma unroll
    for (int nt = 0; nt < 4; ++nt)
#pragma unroll
      for (int i = 0; i < 8; ++i)
        pb_[(i + 8 * half) * 72 + nt * 16 + r] = (__bf16)sc[nt][i];
    asm volatile("s_wait_dscnt 0x0" ::: "memory");
    v16bf ap0 = load_a_frag(pb_, 72, 0, r, half);
    v16bf ap1 = load_a_frag(pb_ + 32, 72, 0, r, half);
#pragma unroll
    for (int dt = 0; dt < 2; ++dt) {
      v16bf bv0 = load_b_frag(vsb, 40, dt * 16, r, half);
      v16bf bv1 = load_b_frag(vsb + (size_t)32 * 40, 40, dt * 16, r, half);
      oacc[dt] = WMMA_BF16(ap0, bv0, oacc[dt]);
      oacc[dt] = WMMA_BF16(ap1, bv1, oacc[dt]);
    }
    if (ch + 1 < 9) { async_wait(); __syncthreads(); }
  }
  // normalize and scatter back to token-major bf16
#pragma unroll
  for (int dt = 0; dt < 2; ++dt)
#pragma unroll
    for (int i = 0; i < 8; ++i) {
      const int qr = wave * 16 + i + 8 * half;
      const int il = qr >> 4, jl = qr & 15;
      const int tok = bb * 16384 + (wr * 16 + il) * 128 + (wc * 16 + jl);
      const float val = oacc[dt][i] / lrow[i];
      obf[(size_t)tok * DIM + head * HD + dt * 16 + r] = (__bf16)val;
    }
}

// ---------------------------------------------------------------------------
extern "C" void kernel_launch(void* const* d_in, const int* in_sizes, int n_in,
                              void* d_out, int out_size, void* d_ws, size_t ws_size,
                              hipStream_t stream) {
  (void)in_sizes; (void)n_in; (void)out_size; (void)ws_size;
  const float* x   = (const float*)d_in[0];
  const float* y   = (const float*)d_in[1];
  const int*   rpi = (const int*)d_in[2];
  const float* n1g = (const float*)d_in[5];
  const float* n1b = (const float*)d_in[6];
  const float* kvw = (const float*)d_in[7];
  const float* kvb = (const float*)d_in[8];
  const float* qw  = (const float*)d_in[9];
  const float* qb  = (const float*)d_in[10];
  const float* rpb = (const float*)d_in[11];
  const float* pw  = (const float*)d_in[12];
  const float* pb  = (const float*)d_in[13];
  const float* n2g = (const float*)d_in[14];
  const float* n2b = (const float*)d_in[15];
  const float* f1w = (const float*)d_in[16];
  const float* f1b = (const float*)d_in[17];
  const float* f2w = (const float*)d_in[18];
  const float* f2b = (const float*)d_in[19];
  float* out = (float*)d_out;

  char* ws = (char*)d_ws;
  const size_t td = (size_t)TOK * DIM * sizeof(__bf16);  // 12.6MB
  __bf16* ln2 = (__bf16*)(ws);
  __bf16* xn  = (__bf16*)(ws + td);
  __bf16* yn  = (__bf16*)(ws + 2 * td);
  __bf16* qv  = (__bf16*)(ws + 3 * td);
  __bf16* kv  = (__bf16*)(ws + 4 * td);        // 2*td (interleaved, ld=384)
  __bf16* ao  = (__bf16*)(ws + 6 * td);
  __bf16* hbf = (__bf16*)(ws + td);            // aliases xn..kv[0:td] (dead by then)
  float*  bias_t = (float*)(ws + 7 * td);      // 6*576*256*4 = 3.5MB
  char*   wb = ws + 7 * td + (size_t)HEADS * 576 * 256 * sizeof(float);
  __bf16* qwb  = (__bf16*)wb;                         // 192*192
  __bf16* kvwb = qwb + DIM * DIM;                     // 192*384
  __bf16* pwb  = kvwb + DIM * 2 * DIM;                // 192*192
  __bf16* f1wb = pwb + DIM * DIM;                     // 192*768
  __bf16* f2wb = f1wb + DIM * MLP_H;                  // 768*192

  dim3 blk(256);
  // one-time precomputation (deterministic, every call)
  f32_to_bf16_kernel<<<(DIM * DIM + 255) / 256, blk, 0, stream>>>(qw, qwb, DIM * DIM);
  f32_to_bf16_kernel<<<(DIM * 2 * DIM + 255) / 256, blk, 0, stream>>>(kvw, kvwb, DIM * 2 * DIM);
  f32_to_bf16_kernel<<<(DIM * DIM + 255) / 256, blk, 0, stream>>>(pw, pwb, DIM * DIM);
  f32_to_bf16_kernel<<<(DIM * MLP_H + 255) / 256, blk, 0, stream>>>(f1w, f1wb, DIM * MLP_H);
  f32_to_bf16_kernel<<<(MLP_H * DIM + 255) / 256, blk, 0, stream>>>(f2w, f2wb, MLP_H * DIM);
  bias_pre_kernel<<<(HEADS * 576 * 256) / 256, blk, 0, stream>>>(rpi, rpb, bias_t);

  // LN1(x), LN1(y) -> bf16
  ln_bf16_kernel<<<TOK / 8, blk, 0, stream>>>(x, n1g, n1b, xn);
  ln_bf16_kernel<<<TOK / 8, blk, 0, stream>>>(y, n1g, n1b, yn);
  // q = yn @ q_w + q_b ; kv = xn @ kv_w + kv_b
  gemm_bf16_kernel<0><<<dim3(TOK / 64, 3), blk, 0, stream>>>(yn, DIM, qwb, DIM, qb, qv, DIM, nullptr, DIM);
  gemm_bf16_kernel<0><<<dim3(TOK / 64, 6), blk, 0, stream>>>(xn, DIM, kvwb, 2 * DIM, kvb, kv, 2 * DIM, nullptr, DIM);
  // windowed attention with relative position bias
  attn_kernel<<<dim3(128, HEADS), dim3(512), 0, stream>>>(qv, kv, bias_t, ao);
  // xo = ao @ proj_w + proj_b + x   (f32, into d_out)
  gemm_bf16_kernel<2><<<dim3(TOK / 64, 3), blk, 0, stream>>>(ao, DIM, pwb, DIM, pb, out, DIM, x, DIM);
  // ln2 = LN2(xo) -> bf16
  ln_bf16_kernel<<<TOK / 8, blk, 0, stream>>>(out, n2g, n2b, ln2);
  // h = gelu(ln2 @ fc1_w + fc1_b) -> bf16
  gemm_bf16_kernel<1><<<dim3(TOK / 64, 12), blk, 0, stream>>>(ln2, DIM, f1wb, MLP_H, f1b, hbf, MLP_H, nullptr, DIM);
  // out = h @ fc2_w + fc2_b + xo
  gemm_bf16_kernel<2><<<dim3(TOK / 64, 3), blk, 0, stream>>>(hbf, MLP_H, f2wb, DIM, f2b, out, DIM, out, MLP_H);
}
